// Discriminator_35588099015245
// MI455X (gfx1250) — matrix operand order repaired and checked by reading, not compiled, not run on hardware
//
#include <hip/hip_runtime.h>

// ---- problem constants (match reference) ----
#define NN 25600     // nodes
#define EE 51200     // edges
#define BB 512       // graphs
#define DD 64        // node hidden
#define EH 128       // edge hidden
#define NA 40
#define NB 10
#define LAYERS 6
#define ITERS 6
#define OUTF 2

typedef __attribute__((ext_vector_type(16))) _Float16 v16h;
typedef __attribute__((ext_vector_type(8)))  _Float16 v8h;
typedef __attribute__((ext_vector_type(8)))  float    v8f;

// ---------------- helpers ----------------
__device__ __forceinline__ float sigmf(float x) { return 1.0f / (1.0f + expf(-x)); }

__device__ __forceinline__ float wsum32(float v) {
#pragma unroll
  for (int m = 16; m >= 1; m >>= 1) v += __shfl_xor(v, m, 32);
  return v;
}

// load A fragment (16x32 f16 tile) in contiguous k order: lane half h holds k0+16h..k0+16h+15 (same as load_b_frag)
__device__ __forceinline__ v16h load_a_frag(const _Float16* __restrict__ A, size_t row,
                                            int K, int k0, int half) {
  const _Float16* p = A + row * (size_t)K + k0 + half * 16;
  v8h lo = *(const v8h*)(p);
  v8h hi = *(const v8h*)(p + 8);
  v16h f;
#pragma unroll
  for (int j = 0; j < 8; ++j) { f[j] = lo[j]; f[8 + j] = hi[j]; }
  return f;
}

// load B fragment (32x16 f16 tile); Bw stored [Ncols, K] row-major (weight [out,in])
__device__ __forceinline__ v16h load_b_frag(const _Float16* __restrict__ Bw, size_t col,
                                            int K, int k0, int half) {
  return *(const v16h*)(Bw + col * (size_t)K + k0 + half * 16);
}

// ---------------- precision convert ----------------
__global__ void k_cvt_f16(const float* __restrict__ in, _Float16* __restrict__ out, int n) {
  int i = blockIdx.x * blockDim.x + threadIdx.x;
  if (i < n) out[i] = (_Float16)in[i];
}

// ---------------- node embedding: h = x_node @ Wᵀ + b ----------------
__global__ void k_node_emb(const float* __restrict__ xn, const float* __restrict__ w,
                           const float* __restrict__ b, float* __restrict__ h) {
  int n = blockIdx.x, d = threadIdx.x;  // block 64
  float s = b[d];
  const float* x = xn + (size_t)n * NA;
  const float* wr = w + (size_t)d * NA;
#pragma unroll
  for (int k = 0; k < NA; ++k) s += x[k] * wr[k];
  h[(size_t)n * DD + d] = s;
}

// ---------------- edge: a = relu((x_edge@Wembᵀ+b)@W1ᵀ+b1)  -> f16 ----------------
__global__ void k_edge_a(const float* __restrict__ xe, const float* __restrict__ wemb,
                         const float* __restrict__ bemb, const float* __restrict__ w1,
                         const float* __restrict__ b1, _Float16* __restrict__ aOut) {
  __shared__ float he[EH];
  int e = blockIdx.x, t = threadIdx.x;  // block 128
  float s = bemb[t];
  const float* x = xe + (size_t)e * NB;
  const float* wr = wemb + (size_t)t * NB;
#pragma unroll
  for (int k = 0; k < NB; ++k) s += x[k] * wr[k];
  he[t] = s;
  __syncthreads();
  float s2 = b1[t];
  const float* w1r = w1 + (size_t)t * EH;
#pragma unroll 8
  for (int k = 0; k < EH; ++k) s2 += he[k] * w1r[k];
  aOut[(size_t)e * EH + t] = (_Float16)fmaxf(s2, 0.0f);
}

// ---------------- big WMMA GEMM: w = a[E,128] @ W2ᵀ[128,4096] + b2 -> f16 ----------------
// one wave per block; each wave: 16 edges x 64 out-features, K=128 (4 k-steps, 16 WMMAs)
__global__ void __launch_bounds__(32)
k_w2_gemm(const _Float16* __restrict__ aF16, const _Float16* __restrict__ w2h,
          const float* __restrict__ b2, _Float16* __restrict__ wOut) {
  int lane = threadIdx.x;
  int half = lane >> 4, lm = lane & 15;
  size_t e0 = (size_t)blockIdx.x * 16;
  int n0 = blockIdx.y * 64;
  v8f acc[4];
#pragma unroll
  for (int t = 0; t < 4; ++t) {
    float bv = b2[n0 + t * 16 + lm];
#pragma unroll
    for (int r = 0; r < 8; ++r) acc[t][r] = bv;
  }
#pragma unroll
  for (int kb = 0; kb < 4; ++kb) {
    int k0 = kb * 32;
    v16h af = load_a_frag(aF16, e0 + lm, EH, k0, half);
#pragma unroll
    for (int t = 0; t < 4; ++t) {
      v16h bf = load_b_frag(w2h, (size_t)(n0 + t * 16 + lm), EH, k0, half);
      acc[t] = __builtin_amdgcn_wmma_f32_16x16x32_f16(false, af, false, bf,
                                                      (short)0, acc[t], false, false);
    }
  }
#pragma unroll
  for (int t = 0; t < 4; ++t)
#pragma unroll
    for (int r = 0; r < 8; ++r) {
      size_t e = e0 + r + half * 8;
      wOut[e * 4096 + n0 + t * 16 + lm] = (_Float16)acc[t][r];
    }
}

// ---------------- generic WMMA GEMM: C[M,Nc] = A[M,K]f16 @ Bw[Nc,K]ᵀf16 (f32 out) ----------------
__global__ void __launch_bounds__(32)
k_gemm16(const _Float16* __restrict__ A, const _Float16* __restrict__ Bw,
         float* __restrict__ C, int K, int Ncols) {
  int lane = threadIdx.x;
  int half = lane >> 4, lm = lane & 15;
  size_t r0 = (size_t)blockIdx.x * 16;
  int n0 = blockIdx.y * 16;
  v8f acc = {};
  for (int k0 = 0; k0 < K; k0 += 32) {
    v16h af = load_a_frag(A, r0 + lm, K, k0, half);
    v16h bf = load_b_frag(Bw, (size_t)(n0 + lm), K, k0, half);
    acc = __builtin_amdgcn_wmma_f32_16x16x32_f16(false, af, false, bf,
                                                 (short)0, acc, false, false);
  }
#pragma unroll
  for (int r = 0; r < 8; ++r)
    C[(r0 + r + half * 8) * (size_t)Ncols + n0 + lm] = acc[r];
}

// ---------------- degree count ----------------
__global__ void k_cnt(const int* __restrict__ dst, float* __restrict__ cnt, int n) {
  int e = blockIdx.x * blockDim.x + threadIdx.x;
  if (e < n) atomicAdd(&cnt[dst[e]], 1.0f);
}

// ---------------- per-edge message: m_e = h[src] @ w_e, scatter-add to msum[dst] --------
// One wave32 per edge. lane L = r*8+c: owns outputs o=c*8..c*8+7, i-block i=r*16..r*16+15.
// w_e (8KB) consumed with 16 global_load_b128 per lane (512B/wave/instr, contiguous).
__global__ void __launch_bounds__(32)
k_msg(const float* __restrict__ h, const _Float16* __restrict__ w,
      const int* __restrict__ src, const int* __restrict__ dst, float* __restrict__ msum) {
  int e = blockIdx.x;
  int lane = threadIdx.x;
  int r = lane >> 3, c = lane & 7;
  const float* hsrc = h + (size_t)src[e] * DD + r * 16;
  float4 hv0 = ((const float4*)hsrc)[0];
  float4 hv1 = ((const float4*)hsrc)[1];
  float4 hv2 = ((const float4*)hsrc)[2];
  float4 hv3 = ((const float4*)hsrc)[3];
  float hs[16] = {hv0.x, hv0.y, hv0.z, hv0.w, hv1.x, hv1.y, hv1.z, hv1.w,
                  hv2.x, hv2.y, hv2.z, hv2.w, hv3.x, hv3.y, hv3.z, hv3.w};
  const _Float16* we = w + (size_t)e * 4096 + (size_t)r * 16 * DD + c * 8;
  float acc[8] = {};
#pragma unroll
  for (int ib = 0; ib < 16; ++ib) {
    v8h wv = *(const v8h*)(we + ib * DD);   // i = r*16+ib, o = c*8..c*8+7
    float hb = hs[ib];
#pragma unroll
    for (int j = 0; j < 8; ++j) acc[j] += hb * (float)wv[j];
  }
#pragma unroll
  for (int j = 0; j < 8; ++j) {
    acc[j] += __shfl_xor(acc[j], 8, 32);
    acc[j] += __shfl_xor(acc[j], 16, 32);
  }
  if (lane < 8) {
    float* mrow = msum + (size_t)dst[e] * DD + c * 8;
#pragma unroll
    for (int j = 0; j < 8; ++j) atomicAdd(&mrow[j], acc[j]);
  }
}

// ---------------- m = relu(msum/cnt) -> f16 ; also hg -> f16 ----------------
__global__ void k_mean_relu(const float* __restrict__ msum, const float* __restrict__ cnt,
                            const float* __restrict__ hg, _Float16* __restrict__ mh,
                            _Float16* __restrict__ hgh) {
  size_t idx = (size_t)blockIdx.x * blockDim.x + threadIdx.x;
  if (idx >= (size_t)NN * DD) return;
  int n = (int)(idx >> 6);
  float c = fmaxf(cnt[n], 1.0f);
  mh[idx] = (_Float16)fmaxf(msum[idx] / c, 0.0f);
  hgh[idx] = (_Float16)hg[idx];
}

// ---------------- GRU gate update (in-place on h) ----------------
__global__ void k_gru_gate(const float* __restrict__ gx, const float* __restrict__ gh,
                           const float* __restrict__ bi, const float* __restrict__ bh,
                           float* __restrict__ hbuf) {
  size_t idx = (size_t)blockIdx.x * blockDim.x + threadIdx.x;
  if (idx >= (size_t)NN * DD) return;
  size_t n = idx >> 6;
  int d = (int)(idx & 63);
  const float* gxr = gx + n * 192;
  const float* ghr = gh + n * 192;
  float r = sigmf(gxr[d] + bi[d] + ghr[d] + bh[d]);
  float z = sigmf(gxr[64 + d] + bi[64 + d] + ghr[64 + d] + bh[64 + d]);
  float nn = tanhf(gxr[128 + d] + bi[128 + d] + r * (ghr[128 + d] + bh[128 + d]));
  float old = hbuf[idx];
  hbuf[idx] = (1.0f - z) * nn + z * old;
}

// ---------------- graph node ranges (node2graph is sorted) ----------------
__global__ void k_range_init(int* __restrict__ gs, int* __restrict__ ge) {
  int g = blockIdx.x * blockDim.x + threadIdx.x;
  if (g < BB) { gs[g] = NN; ge[g] = 0; }
}
__global__ void k_range(const int* __restrict__ n2g, int* __restrict__ gs, int* __restrict__ ge) {
  int n = blockIdx.x * blockDim.x + threadIdx.x;
  if (n >= NN) return;
  int g = n2g[n];
  atomicMin(&gs[g], n);
  atomicMax(&ge[g], n + 1);
}

// ---------------- LSTM: g = x@Wihᵀ + h@Whhᵀ + bih + bhh ----------------
__global__ void __launch_bounds__(256)
k_lstm_mm(const float* __restrict__ x, int in_dim, const float* __restrict__ h,
          const float* __restrict__ wih, const float* __restrict__ whh,
          const float* __restrict__ bih, const float* __restrict__ bhh,
          float* __restrict__ g) {
  int b = blockIdx.x, j = threadIdx.x;  // block 256
  float s = bih[j] + bhh[j];
  const float* xv = x + (size_t)b * in_dim;
  const float* wi = wih + (size_t)j * in_dim;
  for (int k = 0; k < in_dim; ++k) s += xv[k] * wi[k];
  const float* hv = h + (size_t)b * DD;
  const float* wh = whh + (size_t)j * DD;
#pragma unroll 8
  for (int k = 0; k < DD; ++k) s += hv[k] * wh[k];
  g[(size_t)b * 256 + j] = s;
}

__global__ void k_lstm_gate(const float* __restrict__ g, float* __restrict__ h,
                            float* __restrict__ c) {
  int idx = blockIdx.x * blockDim.x + threadIdx.x;
  if (idx >= BB * DD) return;
  int b = idx >> 6, d = idx & 63;
  const float* gr = g + (size_t)b * 256;
  float i = sigmf(gr[d]);
  float f = sigmf(gr[64 + d]);
  float gg = tanhf(gr[128 + d]);
  float o = sigmf(gr[192 + d]);
  float cn = f * c[idx] + i * gg;
  c[idx] = cn;
  h[idx] = o * tanhf(cn);
}

// ---------------- set2set attention + readout (one wave32 per graph) ----------------
__global__ void __launch_bounds__(32)
k_att(const float* __restrict__ h, const float* __restrict__ q,
      const int* __restrict__ gs, const int* __restrict__ ge,
      float* __restrict__ eN, float* __restrict__ qstar) {
  int g = blockIdx.x, lane = threadIdx.x;
  int s = gs[g], e = ge[g];
  float q0 = q[(size_t)g * DD + lane];
  float q1 = q[(size_t)g * DD + lane + 32];
  float mx = -3.0e38f;
  for (int n = s; n < e; ++n) {
    float p = h[(size_t)n * DD + lane] * q0 + h[(size_t)n * DD + lane + 32] * q1;
    p = wsum32(p);            // all lanes hold the full dot product
    eN[n] = p;                // all lanes write the same value
    mx = fmaxf(mx, p);
  }
  __threadfence();
  float se = 0.0f;
  for (int n = s; n < e; ++n) se += expf(eN[n] - mx);
  float inv = se > 0.0f ? 1.0f / se : 0.0f;
  float r0 = 0.0f, r1 = 0.0f;
  for (int n = s; n < e; ++n) {
    float a = expf(eN[n] - mx) * inv;
    r0 += a * h[(size_t)n * DD + lane];
    r1 += a * h[(size_t)n * DD + lane + 32];
  }
  float* qs = qstar + (size_t)g * 2 * DD;
  qs[lane] = q0;  qs[32 + lane] = q1;
  qs[64 + lane] = r0;  qs[96 + lane] = r1;
}

// ---------------- classifier ----------------
__global__ void __launch_bounds__(64)
k_cls(const float* __restrict__ qstar, const float* __restrict__ c1w,
      const float* __restrict__ c1b, const float* __restrict__ c2w,
      const float* __restrict__ c2b, float* __restrict__ out) {
  __shared__ float hid[DD];
  int b = blockIdx.x, t = threadIdx.x;  // block 64
  float s = c1b[t];
  const float* qv = qstar + (size_t)b * 2 * DD;
  const float* wr = c1w + (size_t)t * 2 * DD;
#pragma unroll 8
  for (int k = 0; k < 2 * DD; ++k) s += qv[k] * wr[k];
  hid[t] = fmaxf(s, 0.0f);
  __syncthreads();
  if (t < OUTF) {
    float s2 = c2b[t];
#pragma unroll 8
    for (int j = 0; j < DD; ++j) s2 += hid[j] * c2w[t * DD + j];
    out[(size_t)b * OUTF + t] = s2;
  }
}

// ================= host orchestration =================
extern "C" void kernel_launch(void* const* d_in, const int* in_sizes, int n_in,
                              void* d_out, int out_size, void* d_ws, size_t ws_size,
                              hipStream_t stream) {
  (void)in_sizes; (void)n_in; (void)out_size; (void)ws_size;
  const float* x_node = (const float*)d_in[0];
  const float* x_edge = (const float*)d_in[1];
  int p = 2;
  const float* node_emb_w = (const float*)d_in[p++];
  const float* node_emb_b = (const float*)d_in[p++];
  const float* edge_emb_w = (const float*)d_in[p++];
  const float* edge_emb_b = (const float*)d_in[p++];
  const float* en_w1 = (const float*)d_in[p++];
  const float* en_b1 = (const float*)d_in[p++];
  const float* en_w2 = (const float*)d_in[p++];
  const float* en_b2 = (const float*)d_in[p++];
  const float* gru_wi = (const float*)d_in[p++];
  const float* gru_wh = (const float*)d_in[p++];
  const float* gru_bi = (const float*)d_in[p++];
  const float* gru_bh = (const float*)d_in[p++];
  const float* lstm_wih0 = (const float*)d_in[p++];
  const float* lstm_whh0 = (const float*)d_in[p++];
  const float* lstm_bih0 = (const float*)d_in[p++];
  const float* lstm_bhh0 = (const float*)d_in[p++];
  const float* lstm_wih1 = (const float*)d_in[p++];
  const float* lstm_whh1 = (const float*)d_in[p++];
  const float* lstm_bih1 = (const float*)d_in[p++];
  const float* lstm_bhh1 = (const float*)d_in[p++];
  const float* c1_w = (const float*)d_in[p++];
  const float* c1_b = (const float*)d_in[p++];
  const float* c2_w = (const float*)d_in[p++];
  const float* c2_b = (const float*)d_in[p++];
  const int* src = (const int*)d_in[p++];
  const int* dst = (const int*)d_in[p++];
  const int* n2g = (const int*)d_in[p++];

  // ---- workspace carving ----
  char* base = (char*)d_ws;
  size_t off = 0;
  auto alloc = [&](size_t bytes) -> void* {
    off = (off + 255) & ~(size_t)255;
    void* ptr = base + off;
    off += bytes;
    return ptr;
  };
  _Float16* aF16  = (_Float16*)alloc((size_t)EE * EH * 2);
  _Float16* w2h   = (_Float16*)alloc((size_t)4096 * EH * 2);
  _Float16* wiH   = (_Float16*)alloc((size_t)192 * DD * 2);
  _Float16* whH   = (_Float16*)alloc((size_t)192 * DD * 2);
  _Float16* wEdge = (_Float16*)alloc((size_t)EE * 4096 * 2);   // 419 MB per-edge weights (f16)
  float* hbuf  = (float*)alloc((size_t)NN * DD * 4);
  float* msum  = (float*)alloc((size_t)NN * DD * 4);
  _Float16* mh  = (_Float16*)alloc((size_t)NN * DD * 2);
  _Float16* hgh = (_Float16*)alloc((size_t)NN * DD * 2);
  float* gx   = (float*)alloc((size_t)NN * 192 * 4);
  float* gh   = (float*)alloc((size_t)NN * 192 * 4);
  float* cnt  = (float*)alloc((size_t)NN * 4);
  float* eN   = (float*)alloc((size_t)NN * 4);
  int* gs = (int*)alloc((size_t)BB * 4);
  int* ge = (int*)alloc((size_t)BB * 4);
  float* qstar = (float*)alloc((size_t)BB * 2 * DD * 4);
  float* h0 = (float*)alloc((size_t)BB * DD * 4);
  float* c0 = (float*)alloc((size_t)BB * DD * 4);
  float* h1 = (float*)alloc((size_t)BB * DD * 4);
  float* c1 = (float*)alloc((size_t)BB * DD * 4);
  float* gls = (float*)alloc((size_t)BB * 256 * 4);

  // ---- weight precision conversion (f32 -> f16) ----
  k_cvt_f16<<<(4096 * EH + 255) / 256, 256, 0, stream>>>(en_w2, w2h, 4096 * EH);
  k_cvt_f16<<<(192 * DD + 255) / 256, 256, 0, stream>>>(gru_wi, wiH, 192 * DD);
  k_cvt_f16<<<(192 * DD + 255) / 256, 256, 0, stream>>>(gru_wh, whH, 192 * DD);

  // ---- embeddings ----
  k_node_emb<<<NN, DD, 0, stream>>>(x_node, node_emb_w, node_emb_b, hbuf);
  k_edge_a<<<EE, EH, 0, stream>>>(x_edge, edge_emb_w, edge_emb_b, en_w1, en_b1, aF16);

  // ---- edge network big GEMM (WMMA) ----
  {
    dim3 grid(EE / 16, 4096 / 64);
    k_w2_gemm<<<grid, 32, 0, stream>>>(aF16, w2h, en_b2, wEdge);
  }

  // ---- in-degree counts ----
  hipMemsetAsync(cnt, 0, (size_t)NN * 4, stream);
  k_cnt<<<(EE + 255) / 256, 256, 0, stream>>>(dst, cnt, EE);

  // ---- MPNN layers ----
  for (int l = 0; l < LAYERS; ++l) {
    hipMemsetAsync(msum, 0, (size_t)NN * DD * 4, stream);
    k_msg<<<EE, 32, 0, stream>>>(hbuf, wEdge, src, dst, msum);
    k_mean_relu<<<(NN * DD + 255) / 256, 256, 0, stream>>>(msum, cnt, hbuf, mh, hgh);
    dim3 gg(NN / 16, 192 / 16);
    k_gemm16<<<gg, 32, 0, stream>>>(mh, wiH, gx, DD, 192);
    k_gemm16<<<gg, 32, 0, stream>>>(hgh, whH, gh, DD, 192);
    k_gru_gate<<<(NN * DD + 255) / 256, 256, 0, stream>>>(gx, gh, gru_bi, gru_bh, hbuf);
  }

  // ---- graph node ranges (node2graph is sorted) ----
  k_range_init<<<(BB + 255) / 256, 256, 0, stream>>>(gs, ge);
  k_range<<<(NN + 255) / 256, 256, 0, stream>>>(n2g, gs, ge);

  // ---- Set2Set ----
  hipMemsetAsync(qstar, 0, (size_t)BB * 2 * DD * 4, stream);
  hipMemsetAsync(h0, 0, (size_t)BB * DD * 4, stream);
  hipMemsetAsync(c0, 0, (size_t)BB * DD * 4, stream);
  hipMemsetAsync(h1, 0, (size_t)BB * DD * 4, stream);
  hipMemsetAsync(c1, 0, (size_t)BB * DD * 4, stream);
  for (int it = 0; it < ITERS; ++it) {
    k_lstm_mm<<<BB, 256, 0, stream>>>(qstar, 2 * DD, h0, lstm_wih0, lstm_whh0,
                                      lstm_bih0, lstm_bhh0, gls);
    k_lstm_gate<<<(BB * DD + 255) / 256, 256, 0, stream>>>(gls, h0, c0);
    k_lstm_mm<<<BB, 256, 0, stream>>>(h0, DD, h1, lstm_wih1, lstm_whh1,
                                      lstm_bih1, lstm_bhh1, gls);
    k_lstm_gate<<<(BB * DD + 255) / 256, 256, 0, stream>>>(gls, h1, c1);
    k_att<<<BB, 32, 0, stream>>>(hbuf, h1, gs, ge, eN, qstar);
  }

  // ---- classifier ----
  k_cls<<<BB, DD, 0, stream>>>(qstar, c1_w, c1_b, c2_w, c2_b, (float*)d_out);
}
